// GatedDeltaNet_84774064488856
// MI455X (gfx1250) — compile-verified, matrix operand order checked by reading, not run on hardware
//
#include <hip/hip_runtime.h>

// ---------------------------------------------------------------------------
// GatedDeltaNet forward for MI455X (gfx1250, wave32, WMMA bf16 + async-LDS)
// B=2 T=1024 D=2048 H=8 DK=128 DV=256 KD=1024 VD=2048 KCONV=4
// ---------------------------------------------------------------------------

#define Bc   2
#define Tc   1024
#define Dc   2048
#define Hc   8
#define DKc  128
#define DVc  256
#define KDc  1024
#define VDc  2048
#define Mrows (Bc * Tc)          // 2048 token rows

typedef __attribute__((ext_vector_type(16))) __bf16        v16bf;
typedef __attribute__((ext_vector_type(8)))  float         v8f;
typedef __attribute__((ext_vector_type(8)))  unsigned int  v8u;

// ---------------- f32 -> bf16 (round-to-nearest-even) ----------------------
__device__ __forceinline__ unsigned short f2bf(float f) {
  unsigned int u = __builtin_bit_cast(unsigned int, f);
  u += 0x7FFFu + ((u >> 16) & 1u);
  return (unsigned short)(u >> 16);
}

__global__ __launch_bounds__(256) void to_bf16_kernel(const float* __restrict__ x,
                                                      unsigned short* __restrict__ y,
                                                      int n) {
  int i = blockIdx.x * 256 + threadIdx.x;
  if (i < n) y[i] = f2bf(x[i]);
}

// ---------------- WMMA bf16 GEMM:  C[M,N] = A[M,K] * B[N,K]^T --------------
// Block = 256 threads (8 waves). Wave tile = 32(M) x 64(N); block = 128 x 128.
// 8 WMMAs per 6 fragment loads per K-step of 32.
__device__ __forceinline__ v16bf load_frag_bf16(const unsigned short* __restrict__ p) {
  // p already includes row*K + half*8.  16-bit A/B fragment layout (ISA 7.12.2):
  // VGPR j<4 : K = 2j (+half*8) ; VGPR j>=4 : K = 16 + 2(j-4) (+half*8)
  v8u u;
  u[0] = *(const unsigned int*)(p + 0);
  u[1] = *(const unsigned int*)(p + 2);
  u[2] = *(const unsigned int*)(p + 4);
  u[3] = *(const unsigned int*)(p + 6);
  u[4] = *(const unsigned int*)(p + 16);
  u[5] = *(const unsigned int*)(p + 18);
  u[6] = *(const unsigned int*)(p + 20);
  u[7] = *(const unsigned int*)(p + 22);
  return __builtin_bit_cast(v16bf, u);
}

__global__ __launch_bounds__(256)
void wmma_gemm_bf16(const unsigned short* __restrict__ A,
                    const unsigned short* __restrict__ Bm,
                    float* __restrict__ C, int N, int K) {
  const int lane = threadIdx.x & 31;
  const int wave = threadIdx.x >> 5;
  const int m0 = blockIdx.x * 128 + (wave & 3) * 32;
  const int n0 = blockIdx.y * 128 + (wave >> 2) * 64;
  const int half = lane >> 4;

  const unsigned short* ap0 = A + (size_t)(m0 + (lane & 15)) * K + half * 8;
  const unsigned short* ap1 = ap0 + (size_t)16 * K;
  const unsigned short* bp  = Bm + (size_t)(n0 + (lane & 15)) * K + half * 8;
  const size_t bs = (size_t)16 * K;

  v8f zero = {};
  v8f acc[2][4];
#pragma unroll
  for (int i = 0; i < 2; ++i)
#pragma unroll
    for (int j = 0; j < 4; ++j) acc[i][j] = zero;

  for (int k0 = 0; k0 < K; k0 += 32) {
    __builtin_prefetch(ap0 + k0 + 128, 0, 3);
    __builtin_prefetch(ap1 + k0 + 128, 0, 3);
    __builtin_prefetch(bp + k0 + 128, 0, 3);
    v16bf a0 = load_frag_bf16(ap0 + k0);
    v16bf a1 = load_frag_bf16(ap1 + k0);
#pragma unroll
    for (int j = 0; j < 4; ++j) {
      v16bf b = load_frag_bf16(bp + (size_t)j * bs + k0);
      acc[0][j] = __builtin_amdgcn_wmma_f32_16x16x32_bf16(false, a0, false, b, (short)0, acc[0][j], false, false);
      acc[1][j] = __builtin_amdgcn_wmma_f32_16x16x32_bf16(false, a1, false, b, (short)0, acc[1][j], false, false);
    }
  }
  // C/D layout: element r of v8f -> row +r + 8*half, col n0 + (lane&15)
  const int nn = lane & 15;
#pragma unroll
  for (int i = 0; i < 2; ++i) {
    float* c0 = C + (size_t)(m0 + i * 16 + half * 8) * N + n0 + nn;
#pragma unroll
    for (int r = 0; r < 8; ++r) {
#pragma unroll
      for (int j = 0; j < 4; ++j) c0[(size_t)r * N + j * 16] = acc[i][j][r];
    }
  }
}

// ---------------- a/b head projections (N=8 each, too narrow for WMMA) -----
__global__ __launch_bounds__(256)
void ab_proj_kernel(const float* __restrict__ h, const float* __restrict__ aw,
                    const float* __restrict__ bw, const float* __restrict__ A_log,
                    const float* __restrict__ dt_bias,
                    float* __restrict__ gdec, float* __restrict__ beta) {
  __shared__ float red[16][256];
  const int row = blockIdx.x, tid = threadIdx.x;
  float pa[8] = {0.f, 0.f, 0.f, 0.f, 0.f, 0.f, 0.f, 0.f};
  float pb[8] = {0.f, 0.f, 0.f, 0.f, 0.f, 0.f, 0.f, 0.f};
  const float* hr = h + (size_t)row * Dc;
  for (int d = tid; d < Dc; d += 256) {
    float hv = hr[d];
#pragma unroll
    for (int j = 0; j < 8; ++j) {
      pa[j] = fmaf(hv, aw[j * Dc + d], pa[j]);
      pb[j] = fmaf(hv, bw[j * Dc + d], pb[j]);
    }
  }
#pragma unroll
  for (int j = 0; j < 8; ++j) { red[j][tid] = pa[j]; red[8 + j][tid] = pb[j]; }
  __syncthreads();
  if (tid < 16) {
    float s = 0.f;
    for (int i = 0; i < 256; ++i) s += red[tid][i];
    if (tid < 8) {
      float x  = s + dt_bias[tid];
      float sp = (x > 20.f) ? x : log1pf(expf(x));
      gdec[(size_t)row * Hc + tid] = -expf(A_log[tid]) * sp;
    } else {
      beta[(size_t)row * Hc + (tid - 8)] = 1.f / (1.f + expf(-s));
    }
  }
}

// ---------------- depthwise causal conv(K=4)+SiLU (+L2 norm for q/k) -------
__device__ __forceinline__ float silu_f(float x) { return x / (1.f + expf(-x)); }

__global__ __launch_bounds__(256)
void conv_silu_l2_kernel(const float* __restrict__ pre, const float* __restrict__ cw,
                         float* __restrict__ out) {
  const int gw   = (blockIdx.x * 256 + threadIdx.x) >> 5;   // 0 .. M*H-1
  const int lane = threadIdx.x & 31;
  const int bt = gw / Hc, hh = gw % Hc;
  const int t  = bt % Tc;
  float s[4]; float ss = 0.f;
#pragma unroll
  for (int j = 0; j < 4; ++j) {
    int c = hh * DKc + lane + 32 * j;
    float acc = 0.f;
#pragma unroll
    for (int k = 0; k < 4; ++k) {
      int tk = t - 3 + k;
      if (tk >= 0) acc = fmaf(pre[(size_t)(bt - 3 + k) * KDc + c], cw[c * 4 + k], acc);
    }
    s[j] = silu_f(acc);
    ss = fmaf(s[j], s[j], ss);
  }
#pragma unroll
  for (int off = 16; off; off >>= 1) ss += __shfl_xor(ss, off, 32);
  float rs = rsqrtf(ss + 1e-6f);
#pragma unroll
  for (int j = 0; j < 4; ++j)
    out[(size_t)bt * KDc + hh * DKc + lane + 32 * j] = s[j] * rs;
}

__global__ __launch_bounds__(256)
void conv_silu_v_kernel(const float* __restrict__ pre, const float* __restrict__ cw,
                        float* __restrict__ out) {
  int idx = blockIdx.x * 256 + threadIdx.x;          // over M*VD
  int bt = idx / VDc, c = idx % VDc;
  int t = bt % Tc;
  float acc = 0.f;
#pragma unroll
  for (int k = 0; k < 4; ++k) {
    int tk = t - 3 + k;
    if (tk >= 0) acc = fmaf(pre[(size_t)(bt - 3 + k) * VDc + c], cw[c * 4 + k], acc);
  }
  out[idx] = silu_f(acc);
}

// ---------------- gated delta-rule recurrence (sequential over T) ----------
// One block per (b,h); 512 threads; thread owns a 64-row strip of one V col.
// Next step's q/k/v (512 floats) is double-buffered into LDS with gfx1250
// async-LDS copies (ASYNCcnt / s_wait_asynccnt) so the copy overlaps compute.
__device__ __forceinline__ void async_ld_b32(unsigned int lds_off, const float* g) {
  asm volatile("global_load_async_to_lds_b32 %0, %1, off"
               :: "v"(lds_off), "v"(g)
               : "memory");
}

__device__ __forceinline__ const float* scan_src(const float* __restrict__ qn,
                                                 const float* __restrict__ kn,
                                                 const float* __restrict__ vn,
                                                 size_t rqk, int tid) {
  if (tid < 128) return qn + rqk * DKc + tid;
  if (tid < 256) return kn + rqk * DKc + (tid - 128);
  return vn + rqk * DVc + (tid - 256);
}

__global__ __launch_bounds__(512)
void scan_kernel(const float* __restrict__ qn, const float* __restrict__ kn,
                 const float* __restrict__ vn, const float* __restrict__ gdec,
                 const float* __restrict__ beta, float* __restrict__ o_attn) {
  __shared__ float stage[2][512];          // [0..127]=q [128..255]=k [256..511]=v
  __shared__ float red[512], red2[512];
  __shared__ float dec_s[Tc], beta_s[Tc];
  const int bh = blockIdx.x;               // 0 .. B*H-1
  const int b = bh / Hc, h = bh % Hc;
  const int tid = threadIdx.x;
  const int col = tid & 255, rh = tid >> 8, row0 = rh * 64;
  const float scale = 0.08838834764831845f;   // DK^-0.5

  // hoist per-step scalars: decay = exp(g_t), beta_t
  for (int t = tid; t < Tc; t += 512) {
    size_t r = (size_t)(b * Tc + t) * Hc + h;
    dec_s[t]  = expf(gdec[r]);
    beta_s[t] = beta[r];
  }

  float S[64];
#pragma unroll
  for (int r = 0; r < 64; ++r) S[r] = 0.f;

  // prologue: async-stage t=0 into buffer 0
  {
    size_t rqk0 = (size_t)(b * Tc) * Hc + h;
    async_ld_b32((unsigned int)(unsigned long long)&stage[0][tid],
                 scan_src(qn, kn, vn, rqk0, tid));
  }

  for (int t = 0; t < Tc; ++t) {
    const int p = t & 1;
    const size_t rqk = (size_t)(b * Tc + t) * Hc + h;
    if (t + 1 < Tc) {
      async_ld_b32((unsigned int)(unsigned long long)&stage[p ^ 1][tid],
                   scan_src(qn, kn, vn, rqk + Hc, tid));
      asm volatile("s_wait_asynccnt 0x1" ::: "memory");   // t's copy done
    } else {
      asm volatile("s_wait_asynccnt 0x0" ::: "memory");
    }
    __syncthreads();

    const float decay = dec_s[t], bt_ = beta_s[t];
    const float* qq = &stage[p][0];
    const float* kk = &stage[p][128];
    float pr = 0.f;
#pragma unroll 16
    for (int r = 0; r < 64; ++r) {
      S[r] *= decay;
      pr = fmaf(kk[row0 + r], S[r], pr);
    }
    red[tid] = pr;
    __syncthreads();

    const float kv = red[col] + red[256 + col];
    const float u  = (stage[p][256 + col] - kv) * bt_;
    float po = 0.f;
#pragma unroll 16
    for (int r = 0; r < 64; ++r) {
      S[r] = fmaf(kk[row0 + r], u, S[r]);
      po = fmaf(qq[row0 + r], S[r], po);
    }
    red2[tid] = po;
    __syncthreads();
    if (rh == 0) o_attn[rqk * DVc + col] = (red2[col] + red2[256 + col]) * scale;
  }
}

// ---------------- gated RMSNorm + SiLU gate -> bf16 rows for final GEMM ----
__global__ __launch_bounds__(256)
void rms_gate_kernel(const float* __restrict__ o_attn, const float* __restrict__ gate,
                     const float* __restrict__ onw, unsigned short* __restrict__ out_bf) {
  const int gw   = (blockIdx.x * 256 + threadIdx.x) >> 5;   // (row, head)
  const int lane = threadIdx.x & 31;
  const int row = gw / Hc, hh = gw % Hc;
  const size_t base = ((size_t)row * Hc + hh) * DVc;
  float ov[8]; float ss = 0.f;
#pragma unroll
  for (int j = 0; j < 8; ++j) {
    ov[j] = o_attn[base + lane + 32 * j];
    ss = fmaf(ov[j], ov[j], ss);
  }
#pragma unroll
  for (int off = 16; off; off >>= 1) ss += __shfl_xor(ss, off, 32);
  const float rs = rsqrtf(ss * (1.f / DVc) + 1e-5f);
#pragma unroll
  for (int j = 0; j < 8; ++j) {
    int dv = lane + 32 * j;
    float gv = gate[(size_t)row * VDc + hh * DVc + dv];
    float y  = ov[j] * rs * onw[dv] * silu_f(gv);
    out_bf[(size_t)row * VDc + hh * DVc + dv] = f2bf(y);
  }
}

// ---------------------------------------------------------------------------
extern "C" void kernel_launch(void* const* d_in, const int* in_sizes, int n_in,
                              void* d_out, int out_size, void* d_ws, size_t ws_size,
                              hipStream_t stream) {
  const float* h_f   = (const float*)d_in[0];
  const float* q_w   = (const float*)d_in[1];
  const float* k_w   = (const float*)d_in[2];
  const float* v_w   = (const float*)d_in[3];
  const float* a_w   = (const float*)d_in[4];
  const float* b_w   = (const float*)d_in[5];
  const float* g_w   = (const float*)d_in[6];
  const float* o_w   = (const float*)d_in[7];
  const float* qcw   = (const float*)d_in[8];
  const float* kcw   = (const float*)d_in[9];
  const float* vcw   = (const float*)d_in[10];
  const float* A_log = (const float*)d_in[11];
  const float* dtb   = (const float*)d_in[12];
  const float* onw   = (const float*)d_in[13];

  size_t off = 0;
  auto take = [&](size_t bytes) -> void* {
    void* p = (char*)d_ws + off;
    off += (bytes + 255) & ~(size_t)255;
    return p;
  };
  unsigned short* h_bf  = (unsigned short*)take((size_t)Mrows * Dc * 2);
  unsigned short* wq_bf = (unsigned short*)take((size_t)KDc * Dc * 2);
  unsigned short* wk_bf = (unsigned short*)take((size_t)KDc * Dc * 2);
  unsigned short* wv_bf = (unsigned short*)take((size_t)VDc * Dc * 2);
  unsigned short* wg_bf = (unsigned short*)take((size_t)VDc * Dc * 2);
  unsigned short* wo_bf = (unsigned short*)take((size_t)Dc * VDc * 2);
  float* q_pre  = (float*)take((size_t)Mrows * KDc * 4);
  float* k_pre  = (float*)take((size_t)Mrows * KDc * 4);
  float* v_pre  = (float*)take((size_t)Mrows * VDc * 4);
  float* gate   = (float*)take((size_t)Mrows * VDc * 4);
  float* qn     = (float*)take((size_t)Mrows * KDc * 4);
  float* kn     = (float*)take((size_t)Mrows * KDc * 4);
  float* vn     = (float*)take((size_t)Mrows * VDc * 4);
  float* gdec   = (float*)take((size_t)Mrows * Hc * 4);
  float* betab  = (float*)take((size_t)Mrows * Hc * 4);
  float* o_attn = (float*)take((size_t)Mrows * VDc * 4);
  unsigned short* o_bf = (unsigned short*)take((size_t)Mrows * VDc * 2);

  const dim3 blk(256);
  // f32 -> bf16 operand staging
  to_bf16_kernel<<<(Mrows * Dc) / 256, blk, 0, stream>>>(h_f, h_bf, Mrows * Dc);
  to_bf16_kernel<<<(KDc * Dc) / 256, blk, 0, stream>>>(q_w, wq_bf, KDc * Dc);
  to_bf16_kernel<<<(KDc * Dc) / 256, blk, 0, stream>>>(k_w, wk_bf, KDc * Dc);
  to_bf16_kernel<<<(VDc * Dc) / 256, blk, 0, stream>>>(v_w, wv_bf, VDc * Dc);
  to_bf16_kernel<<<(VDc * Dc) / 256, blk, 0, stream>>>(g_w, wg_bf, VDc * Dc);
  to_bf16_kernel<<<(Dc * VDc) / 256, blk, 0, stream>>>(o_w, wo_bf, Dc * VDc);

  // WMMA projections: C[M,N] = A[M,K] * W[N,K]^T  (block tile 128x128)
  wmma_gemm_bf16<<<dim3(Mrows / 128, KDc / 128), blk, 0, stream>>>(h_bf, wq_bf, q_pre, KDc, Dc);
  wmma_gemm_bf16<<<dim3(Mrows / 128, KDc / 128), blk, 0, stream>>>(h_bf, wk_bf, k_pre, KDc, Dc);
  wmma_gemm_bf16<<<dim3(Mrows / 128, VDc / 128), blk, 0, stream>>>(h_bf, wv_bf, v_pre, VDc, Dc);
  wmma_gemm_bf16<<<dim3(Mrows / 128, VDc / 128), blk, 0, stream>>>(h_bf, wg_bf, gate, VDc, Dc);

  // narrow head projections (g decay / beta)
  ab_proj_kernel<<<Mrows, blk, 0, stream>>>(h_f, a_w, b_w, A_log, dtb, gdec, betab);

  // conv + SiLU (+ per-head L2 norm for q/k)
  conv_silu_l2_kernel<<<(Mrows * Hc * 32) / 256, blk, 0, stream>>>(q_pre, qcw, qn);
  conv_silu_l2_kernel<<<(Mrows * Hc * 32) / 256, blk, 0, stream>>>(k_pre, kcw, kn);
  conv_silu_v_kernel<<<(Mrows * VDc) / 256, blk, 0, stream>>>(v_pre, vcw, vn);

  // sequential gated delta-rule scan: one block per (b,h), async-LDS staged
  scan_kernel<<<Bc * Hc, 512, 0, stream>>>(qn, kn, vn, gdec, betab, o_attn);

  // gated RMSNorm -> bf16 rows
  rms_gate_kernel<<<(Mrows * Hc * 32) / 256, blk, 0, stream>>>(o_attn, gate, onw, o_bf);

  // final output projection: out[M,D] = o_bf[M,VD] * o_w[D,VD]^T
  wmma_gemm_bf16<<<dim3(Mrows / 128, Dc / 128), blk, 0, stream>>>(o_bf, wo_bf, (float*)d_out, Dc, VDc);
}
